// AdaptiveDeformConv2d_73194832658528
// MI455X (gfx1250) — matrix operand order repaired and checked by reading, not compiled, not run on hardware
//
#include <hip/hip_runtime.h>

typedef __attribute__((ext_vector_type(16))) __bf16 v16bf;
typedef __attribute__((ext_vector_type(8)))  float  v8f;

#define Kc    15
#define K2c   225
#define PIX   2048        // B*H*W
#define Cc    64
#define NOFF  1800        // G*K2*2
#define NMSK  900         // G*K2

__device__ __forceinline__ float silu_f(float x) { return x / (1.0f + expf(-x)); }

// ---------------------------------------------------------------------------
// zero the two scalar outputs (d_out is poisoned once, never re-poisoned)
// ---------------------------------------------------------------------------
__global__ void init_scalars_kernel(float* out) {
  if (threadIdx.x == 0) { out[PIX * Cc] = 0.0f; out[PIX * Cc + 1] = 0.0f; }
}

// ---------------------------------------------------------------------------
// env (gaussian envelope, normalized) + position-conditioned kernel MLP
// ---------------------------------------------------------------------------
__global__ void prep_kernel(const float* raw_sigma,
                            const float* w_k1, const float* b_k1,
                            const float* w_k2, const float* b_k2,
                            const float* w_k3, const float* b_k3,
                            float* env, float* kwf) {
  __shared__ float ssum;
  int t = threadIdx.x;
  if (t == 0) ssum = 0.0f;
  __syncthreads();

  float sp    = log1pf(expf(raw_sigma[0]));
  float sigma = fminf(fmaxf(sp, 1e-3f), 0.5f);

  float e = 0.0f, gh = 0.0f, gw = 0.0f;
  if (t < K2c) {
    gh = -0.5f + (float)(t / Kc) * (1.0f / 14.0f);
    gw = -0.5f + (float)(t % Kc) * (1.0f / 14.0f);
    e  = expf(-(gh * gh + gw * gw) / (2.0f * sigma * sigma));
    atomicAdd(&ssum, e);
  }
  __syncthreads();

  if (t < K2c) {
    env[t] = e / fmaxf(ssum, 1e-8f);

    float p0 = 2.0f * gh, p1 = 2.0f * gw;
    float h1[32], h2[32];
    #pragma unroll
    for (int r = 0; r < 32; ++r)
      h1[r] = silu_f(p0 * w_k1[r] + p1 * w_k1[32 + r] + b_k1[r]);
    for (int r = 0; r < 32; ++r) {
      float a = b_k2[r];
      #pragma unroll
      for (int s = 0; s < 32; ++s) a += h1[s] * w_k2[s * 32 + r];
      h2[r] = silu_f(a);
    }
    for (int n = 0; n < 64; ++n) {
      float a = b_k3[n];
      #pragma unroll
      for (int s = 0; s < 32; ++s) a += h2[s] * w_k3[s * 64 + n];
      kwf[t * 64 + n] = a;   // flat layout; reshape quirk handled at use site
    }
  }
}

// ---------------------------------------------------------------------------
// depthwise 3x3 (SAME, zero pad) + bias + SiLU   -> tbuf [2048,64]
// ---------------------------------------------------------------------------
__global__ void dwconv_silu_kernel(const float* __restrict__ x,
                                   const float* __restrict__ w_dw,
                                   const float* __restrict__ b_dw,
                                   float* __restrict__ t) {
  int idx = blockIdx.x * blockDim.x + threadIdx.x;
  if (idx >= PIX * Cc) return;
  int c  = idx & 63;
  int pw = (idx >> 6) & 31;
  int ph = (idx >> 11) & 31;
  int pb = idx >> 16;
  float acc = b_dw[c];
  #pragma unroll
  for (int dh = -1; dh <= 1; ++dh) {
    #pragma unroll
    for (int dw = -1; dw <= 1; ++dw) {
      int hh = ph + dh, ww = pw + dw;
      if (hh >= 0 && hh < 32 && ww >= 0 && ww < 32)
        acc += x[((pb * 32 + hh) * 32 + ww) * 64 + c] *
               w_dw[((dh + 1) * 3 + (dw + 1)) * 64 + c];
    }
  }
  t[idx] = silu_f(acc);
}

// ---------------------------------------------------------------------------
// WMMA GEMM: C[M,N] = (A[M,64] * rowscale?) @ W[64,N] + bias, * (*postscale)?
// 256 threads = 8 waves per block; block tile = 32 M-rows x 128 N-cols.
// A (32x64 f32) staged once in LDS, shared by all 8 waves; each wave owns a
// 16-col strip and computes two 16x16 M-subtiles -> 4 WMMAs/wave.
// B fragments loaded straight from global (coalesced across lanes).
// Fragment layouts (wave32; D as in CDNA5 ISA 7.12.2):
//   A 16x32 bf16 : lanes 0-15 hold K=0..15, lanes 16-31 hold K=16..31 (this code's order)
//   B 32x16 bf16 : lanes 0-15 hold K=0..15, lanes 16-31 hold K=16..31 (same order as A; not the ISA table order)
//   D 16x16 f32  : VGPR v <-> M = v + 8*(lane/16), N = lane%16
// ---------------------------------------------------------------------------
__global__ void gemm64_wmma_kernel(const float* __restrict__ A,
                                   const float* __restrict__ W,
                                   const float* __restrict__ bias,
                                   float* __restrict__ Cout,
                                   int N, const float* postscale,
                                   const float* ascale) {
  __shared__ __align__(16) float sA[32][64];
  const int tid  = threadIdx.x;          // 256 threads
  const int wave = tid >> 5;
  const int lane = tid & 31;
  const int m0 = blockIdx.x * 32;
  const int n0 = blockIdx.y * 128 + wave * 16;

  // stage A (32x64 f32 = 512 float4), optional per-(batch,channel) scale
  for (int i = tid; i < 512; i += 256) {
    int r = i >> 4, c4 = i & 15;
    float4 v = ((const float4*)(A + (size_t)(m0 + r) * 64))[c4];
    if (ascale) {
      const float* sc = ascale + (((m0 + r) >> 10) * 64) + c4 * 4;
      v.x *= sc[0]; v.y *= sc[1]; v.z *= sc[2]; v.w *= sc[3];
    }
    ((float4*)sA[r])[c4] = v;
  }
  __syncthreads();

  const int hl = lane >> 4;   // lane half
  const int ml = lane & 15;
  const int n  = n0 + ml;
  const bool nok = (n < N);

  // B fragments direct from global: lanes span 16 contiguous columns
  v16bf b0, b1;
  #pragma unroll
  for (int v = 0; v < 8; ++v) {
    int kb = hl * 16 + 2 * v;
    b0[2 * v]     = (__bf16)(nok ? W[(size_t)kb * N + n]        : 0.0f);
    b0[2 * v + 1] = (__bf16)(nok ? W[(size_t)(kb + 1) * N + n]  : 0.0f);
    b1[2 * v]     = (__bf16)(nok ? W[(size_t)(kb + 32) * N + n] : 0.0f);
    b1[2 * v + 1] = (__bf16)(nok ? W[(size_t)(kb + 33) * N + n] : 0.0f);
  }

  float ps = postscale ? postscale[0] : 1.0f;
  float bn = (bias && nok) ? bias[n] : 0.0f;

  #pragma unroll
  for (int mh = 0; mh < 2; ++mh) {
    v16bf a0, a1;
    #pragma unroll
    for (int v = 0; v < 8; ++v) {
      int k = hl * 16 + 2 * v;
      a0[2 * v]     = (__bf16)sA[mh * 16 + ml][k];
      a0[2 * v + 1] = (__bf16)sA[mh * 16 + ml][k + 1];
      a1[2 * v]     = (__bf16)sA[mh * 16 + ml][32 + k];
      a1[2 * v + 1] = (__bf16)sA[mh * 16 + ml][33 + k];
    }
    v8f acc = {0.f, 0.f, 0.f, 0.f, 0.f, 0.f, 0.f, 0.f};
    acc = __builtin_amdgcn_wmma_f32_16x16x32_bf16(false, a0, false, b0,
                                                  (short)0, acc, false, false);
    acc = __builtin_amdgcn_wmma_f32_16x16x32_bf16(false, a1, false, b1,
                                                  (short)0, acc, false, false);
    if (nok) {
      #pragma unroll
      for (int v = 0; v < 8; ++v) {
        int m = m0 + mh * 16 + v + 8 * hl;
        Cout[(size_t)m * N + n] = (acc[v] + bn) * ps;
      }
    }
  }
}

// ---------------------------------------------------------------------------
// fused env-multiply + softmax over K2=225 (one wave per (pixel,g) row),
// in-place, plus entropy-term accumulation
// ---------------------------------------------------------------------------
__global__ void softmax_env_kernel(float* __restrict__ mask,
                                   const float* __restrict__ env,
                                   float* ent_out) {
  int row  = blockIdx.x * 8 + (threadIdx.x >> 5);   // 8192 rows total
  int lane = threadIdx.x & 31;
  float* mr = mask + (size_t)(row >> 2) * NMSK + (row & 3) * K2c;

  float vals[8];
  float mx = -1e30f;
  int cnt = 0;
  for (int j = lane; j < K2c; j += 32) {
    float v = mr[j] * env[j];
    vals[cnt++] = v;
    mx = fmaxf(mx, v);
  }
  for (int o = 16; o > 0; o >>= 1) mx = fmaxf(mx, __shfl_xor(mx, o, 32));

  float sum = 0.0f;
  cnt = 0;
  for (int j = lane; j < K2c; j += 32) {
    float e = expf(vals[cnt] - mx);
    vals[cnt++] = e;
    sum += e;
  }
  for (int o = 16; o > 0; o >>= 1) sum += __shfl_xor(sum, o, 32);
  float inv = 1.0f / sum;

  float ent = 0.0f;
  cnt = 0;
  for (int j = lane; j < K2c; j += 32) {
    float a = vals[cnt++] * inv;
    mr[j] = a;
    ent += a * logf(a + 1e-8f);
  }
  for (int o = 16; o > 0; o >>= 1) ent += __shfl_xor(ent, o, 32);
  if (lane == 0) atomicAdd(ent_out, ent * (1.0f / 8192.0f));  // = -entropy
}

// ---------------------------------------------------------------------------
// mean of squares (offset regularizer)
// ---------------------------------------------------------------------------
__global__ void sq_reduce_kernel(const float* __restrict__ v, long long n,
                                 float* out, float scale) {
  long long i      = (long long)blockIdx.x * blockDim.x + threadIdx.x;
  long long stride = (long long)gridDim.x * blockDim.x;
  float acc = 0.0f;
  for (; i < n; i += stride) { float x = v[i]; acc += x * x; }
  for (int o = 16; o > 0; o >>= 1) acc += __shfl_xor(acc, o, 32);
  if ((threadIdx.x & 31) == 0) atomicAdd(out, acc * scale);
}

// ---------------------------------------------------------------------------
// deformable bilinear sampling + kernel-weight + attention accumulation
// one thread per (pixel, g, gc) output element; 225 taps each
// ---------------------------------------------------------------------------
__global__ void sample_kernel(const float* __restrict__ xproj,
                              const float* __restrict__ offs,
                              const float* __restrict__ attn,
                              const float* __restrict__ kwf,
                              float* __restrict__ outbuf) {
  int idx = blockIdx.x * blockDim.x + threadIdx.x;
  if (idx >= PIX * Cc) return;
  int c  = idx & 63;
  int gc = c & 15, g = c >> 4;
  int pixel = idx >> 6;
  int pw = pixel & 31, ph = (pixel >> 5) & 31, pb = pixel >> 10;

  const float* orow  = offs + (size_t)pixel * NOFF + g * (K2c * 2);
  const float* arow  = attn + (size_t)pixel * NMSK + g * K2c;
  const float* xbase = xproj + (size_t)pb * 1024 * 64;

  float acc = 0.0f;
  int jr = 0, jc = 0;
  for (int j = 0; j < K2c; ++j) {
    float oh = orow[2 * j], ow = orow[2 * j + 1];
    float ah = (float)ph + (float)(jr - 7) + oh;
    float aw = (float)pw + (float)(jc - 7) + ow;
    float valid = (ah < 0.f || ah > 31.f || aw < 0.f || aw > 31.f) ? 0.f : 1.f;
    float ahc = fminf(fmaxf(ah, 0.f), 31.f);
    float awc = fminf(fmaxf(aw, 0.f), 31.f);
    int hf = (int)floorf(ahc); hf = hf < 0 ? 0 : (hf > 31 ? 31 : hf);
    int wf = (int)floorf(awc); wf = wf < 0 ? 0 : (wf > 31 ? 31 : wf);
    int hcp = hf + 1 > 31 ? 31 : hf + 1;
    int wcp = wf + 1 > 31 ? 31 : wf + 1;
    float fh = ahc - (float)hf, fw = awc - (float)wf;

    const float* r0 = xbase + (size_t)(hf * 32) * 64;
    const float* r1 = xbase + (size_t)(hcp * 32) * 64;
    float vff = r0[wf * 64 + c],  vfc = r0[wcp * 64 + c];
    float vcf = r1[wf * 64 + c],  vcc = r1[wcp * 64 + c];
    float s = vff * (1.f - fh) * (1.f - fw) + vfc * (1.f - fh) * fw +
              vcf * fh * (1.f - fw)         + vcc * fh * fw;

    // torch view quirk: kw(g,j,gc) = kw_raw.flat[g*3600 + j*16 + gc]
    acc += s * valid * kwf[g * 3600 + j * 16 + gc] * arow[j];
    if (++jc == 15) { jc = 0; ++jr; }
  }
  outbuf[idx] = acc;
}

// ---------------------------------------------------------------------------
// SE block: per-batch global pool -> fc1 -> SiLU -> fc2 -> sigmoid scale
// ---------------------------------------------------------------------------
__global__ void se_kernel(const float* __restrict__ outbuf,
                          const float* w_fc1, const float* b_fc1,
                          const float* w_fc2, const float* b_fc2,
                          float* scalebuf) {
  __shared__ float sp[2][64];
  __shared__ float sh[2][16];
  int t = threadIdx.x;          // 128 threads
  int b = t >> 6, c = t & 63;
  float s = 0.0f;
  const float* base = outbuf + (size_t)b * 1024 * 64 + c;
  for (int p = 0; p < 1024; ++p) s += base[(size_t)p * 64];
  sp[b][c] = s * (1.0f / 1024.0f);
  __syncthreads();
  if (t < 32) {
    int bb = t >> 4, r = t & 15;
    float a = b_fc1[r];
    #pragma unroll
    for (int cc = 0; cc < 64; ++cc) a += sp[bb][cc] * w_fc1[cc * 16 + r];
    sh[bb][r] = silu_f(a);
  }
  __syncthreads();
  float a = b_fc2[c];
  #pragma unroll
  for (int r = 0; r < 16; ++r) a += sh[b][r] * w_fc2[r * 64 + c];
  scalebuf[t] = 1.0f / (1.0f + expf(-a));
}

// ---------------------------------------------------------------------------
extern "C" void kernel_launch(void* const* d_in, const int* in_sizes, int n_in,
                              void* d_out, int out_size, void* d_ws, size_t ws_size,
                              hipStream_t stream) {
  const float* x     = (const float*)d_in[0];
  const float* rsig  = (const float*)d_in[1];
  const float* bos   = (const float*)d_in[2];
  const float* w_ip  = (const float*)d_in[3];
  const float* b_ip  = (const float*)d_in[4];
  const float* w_op  = (const float*)d_in[5];
  const float* b_op  = (const float*)d_in[6];
  const float* w_dw  = (const float*)d_in[7];
  const float* b_dw  = (const float*)d_in[8];
  const float* w_pw  = (const float*)d_in[9];
  const float* b_pw  = (const float*)d_in[10];
  const float* w_off = (const float*)d_in[11];
  const float* b_off = (const float*)d_in[12];
  const float* w_msk = (const float*)d_in[13];
  const float* b_msk = (const float*)d_in[14];
  const float* w_k1  = (const float*)d_in[15];
  const float* b_k1  = (const float*)d_in[16];
  const float* w_k2  = (const float*)d_in[17];
  const float* b_k2  = (const float*)d_in[18];
  const float* w_k3  = (const float*)d_in[19];
  const float* b_k3  = (const float*)d_in[20];
  const float* w_fc1 = (const float*)d_in[21];
  const float* b_fc1 = (const float*)d_in[22];
  const float* w_fc2 = (const float*)d_in[23];
  const float* b_fc2 = (const float*)d_in[24];
  float* out = (float*)d_out;

  float* ws     = (float*)d_ws;
  float* env    = ws;                               // 225 (pad 256)
  float* kwf    = ws + 256;                         // 14400
  float* tbuf   = ws + 16384;                       // 131072
  float* xproj  = tbuf + PIX * Cc;                  // 131072
  float* xdw    = xproj + PIX * Cc;                 // 131072
  float* offs   = xdw + PIX * Cc;                   // 2048*1800
  float* attnb  = offs + (size_t)PIX * NOFF;        // 2048*900 (mask->attn in place)
  float* outbuf = attnb + (size_t)PIX * NMSK;       // 131072
  float* scaleb = outbuf + PIX * Cc;                // 128

  init_scalars_kernel<<<1, 32, 0, stream>>>(out);
  prep_kernel<<<1, 256, 0, stream>>>(rsig, w_k1, b_k1, w_k2, b_k2, w_k3, b_k3,
                                     env, kwf);
  dwconv_silu_kernel<<<512, 256, 0, stream>>>(x, w_dw, b_dw, tbuf);

  dim3 g1(PIX / 32, 1);                              // N = 64 GEMMs
  gemm64_wmma_kernel<<<g1, 256, 0, stream>>>(x,    w_ip, b_ip, xproj, 64,
                                             nullptr, nullptr);
  gemm64_wmma_kernel<<<g1, 256, 0, stream>>>(tbuf, w_pw, b_pw, xdw, 64,
                                             nullptr, nullptr);

  dim3 g2(PIX / 32, (NOFF + 127) / 128);             // offsets, scaled in-GEMM
  gemm64_wmma_kernel<<<g2, 256, 0, stream>>>(xdw, w_off, b_off, offs, NOFF,
                                             bos, nullptr);

  dim3 g3(PIX / 32, (NMSK + 127) / 128);             // mask logits
  gemm64_wmma_kernel<<<g3, 256, 0, stream>>>(xdw, w_msk, b_msk, attnb, NMSK,
                                             nullptr, nullptr);

  softmax_env_kernel<<<1024, 256, 0, stream>>>(attnb, env, out + PIX * Cc + 1);
  sq_reduce_kernel<<<256, 256, 0, stream>>>(offs, (long long)PIX * NOFF,
                                            out + PIX * Cc,
                                            1.0f / (float)((long long)PIX * NOFF));
  sample_kernel<<<512, 256, 0, stream>>>(xproj, offs, attnb, kwf, outbuf);
  se_kernel<<<1, 128, 0, stream>>>(outbuf, w_fc1, b_fc1, w_fc2, b_fc2, scaleb);

  // final projection with fused SE channel scaling on A
  gemm64_wmma_kernel<<<g1, 256, 0, stream>>>(outbuf, w_op, b_op, out, 64,
                                             nullptr, scaleb);
}
